// PCT_77824807404164
// MI455X (gfx1250) — matrix operand order repaired and checked by reading, not compiled, not run on hardware
//
#include <hip/hip_runtime.h>
#include <hip/hip_bf16.h>
#include <stdint.h>

// ---------------------------------------------------------------------------
// PCT attention block for MI455X (gfx1250, wave32, WMMA).
// All GEMMs run on v_wmma_f32_16x16x32_bf16 (bf16 operands, fp32 accum).
// Energy + relative-position bias + softmax fused per 16-row block in LDS.
// attn.V GEMM register-blocked 16x64 per wave (A-operand reuse x4).
// ---------------------------------------------------------------------------

#define BB   16
#define CC   256
#define NN   1024
#define DQK  64
#define DV   256
#define RB   64      // 2*BINS-1 = 63, padded to 64

typedef __bf16 bf16;
typedef __attribute__((ext_vector_type(16))) __bf16 v16bf;
typedef __attribute__((ext_vector_type(8)))  float  v8f;
typedef __attribute__((ext_vector_type(4)))  unsigned int u32x4;

union BfReg { u32x4 q[2]; v16bf v; };

// A-matrix (16x32 bf16) per-lane load: row ptr already at (row*stride + kBase).
// lane half kh: VGPR0-3 hold K = kh*8..kh*8+7, VGPR4-7 hold K = 16+kh*8..+7.
__device__ __forceinline__ v16bf ldA(const bf16* p, int kh) {
  BfReg r;
  r.q[0] = *reinterpret_cast<const u32x4*>(p + kh * 8);
  r.q[1] = *reinterpret_cast<const u32x4*>(p + 16 + kh * 8);
  return r.v;
}
// B-matrix (32x16 bf16) per-lane load: K = kh*8..+7, then 16+kh*8..+7.
__device__ __forceinline__ v16bf ldB(const bf16* p, int kh) {
  BfReg r;
  r.q[0] = *reinterpret_cast<const u32x4*>(p + kh * 8);
  r.q[1] = *reinterpret_cast<const u32x4*>(p + 16 + kh * 8);
  return r.v;
}
__device__ __forceinline__ v8f wmma_bf(v16bf a, v16bf b, v8f c) {
  return __builtin_amdgcn_wmma_f32_16x16x32_bf16(false, a, false, b,
                                                 (short)0, c, false, false);
}

// --------------------------- weight packing --------------------------------
__global__ void k_pack(const float* qw, const float* kw, const float* vw,
                       const float* tw, const float* xlt, const float* ylt,
                       const float* zlt, bf16* wqkv, bf16* wtr, bf16* ltb) {
  int i = blockIdx.x * blockDim.x + threadIdx.x;
  const int n0 = 384 * 256, n1 = n0 + 256 * 256, n2 = n1 + 3 * 64 * 64;
  if (i >= n2) return;
  if (i < n0) {
    int row = i >> 8, c = i & 255;
    float v;
    if (row < 64)       v = qw[row * 256 + c];
    else if (row < 128) v = kw[(row - 64) * 256 + c];
    else                v = vw[(row - 128) * 256 + c];
    wqkv[i] = (bf16)v;
  } else if (i < n1) {
    int j = i - n0;
    wtr[j] = (bf16)tw[j];
  } else {
    int j = i - n1;
    int tab = j >> 12, rem = j & 4095;
    int r = rem >> 6, d = rem & 63;
    const float* lt = (tab == 0) ? xlt : (tab == 1 ? ylt : zlt);
    ltb[j] = (bf16)((r < 63) ? lt[r * 64 + d] : 0.0f);
  }
}

// ------------------- x: [B,C,N] fp32 -> xT: [B,N,C] bf16 -------------------
__global__ void k_xT(const float* x, bf16* xT) {
  int i = blockIdx.x * blockDim.x + threadIdx.x;   // b*N + n
  if (i >= BB * NN) return;
  int b = i >> 10, n = i & 1023;
  const float* src = x + (size_t)b * CC * NN + n;
  bf16* dst = xT + (size_t)i * CC;
  for (int c = 0; c < CC; ++c) dst[c] = (bf16)src[(size_t)c * NN];
}

// ---------------- fused QKV projection: [384,256] @ [256,N] ----------------
__global__ void __launch_bounds__(256)
k_proj(const bf16* wqkv, const bf16* xT, const float* vb,
       bf16* xq, bf16* xk, bf16* xv) {
  int t = blockIdx.x * 8 + (threadIdx.x >> 5);
  int l = threadIdx.x & 31;
  int b = t / (24 * 64), rem = t % (24 * 64);
  int rt = rem >> 6, nt = rem & 63;
  int kh = l >> 4, lm = l & 15;
  const bf16* arow = wqkv + (size_t)(rt * 16 + lm) * 256;
  const bf16* brow = xT + ((size_t)b * NN + nt * 16 + lm) * 256;
  v8f acc = {};
#pragma unroll
  for (int kk = 0; kk < 8; ++kk)
    acc = wmma_bf(ldA(arow + kk * 32, kh), ldB(brow + kk * 32, kh), acc);
  int n = nt * 16 + lm;
  if (rt < 4) {            // q rows -> xq[b][n][d]
#pragma unroll
    for (int i = 0; i < 8; ++i) {
      int d = rt * 16 + kh * 8 + i;
      xq[((size_t)b * NN + n) * DQK + d] = (bf16)acc[i];
    }
  } else if (rt < 8) {     // k rows -> xk[b][n][d] (k-per-point, K-major)
#pragma unroll
    for (int i = 0; i < 8; ++i) {
      int d = rt * 16 + kh * 8 + i - 64;
      xk[((size_t)b * NN + n) * DQK + d] = (bf16)acc[i];
    }
  } else {                 // v rows (+bias) -> xv[b][d][n]
#pragma unroll
    for (int i = 0; i < 8; ++i) {
      int d = rt * 16 + kh * 8 + i - 128;
      xv[((size_t)b * DV + d) * NN + n] = (bf16)(acc[i] + vb[d]);
    }
  }
}

// ------------------- ql[tab][b][n][r] = q[n,:] . lt[r,:] -------------------
__global__ void __launch_bounds__(256)
k_ql(const bf16* xq, const bf16* ltb, float* ql) {
  int t = blockIdx.x * 8 + (threadIdx.x >> 5);
  int l = threadIdx.x & 31;
  int tab = t >> 12, rem = t & 4095;
  int b = rem >> 8, rem2 = rem & 255;
  int rt = rem2 >> 6, nt = rem2 & 63;
  int kh = l >> 4, lm = l & 15;
  const bf16* arow = xq + ((size_t)b * NN + nt * 16 + lm) * DQK;
  const bf16* brow = ltb + (size_t)tab * 4096 + (size_t)(rt * 16 + lm) * 64;
  v8f acc = {};
#pragma unroll
  for (int kk = 0; kk < 2; ++kk)
    acc = wmma_bf(ldA(arow + kk * 32, kh), ldB(brow + kk * 32, kh), acc);
#pragma unroll
  for (int i = 0; i < 8; ++i) {
    int n = nt * 16 + kh * 8 + i;
    int r = rt * 16 + lm;
    ql[(((size_t)tab * BB + b) * NN + n) * RB + r] = acc[i];
  }
}

// ------------- fused energy + rel-bias gather + softmax (LDS) --------------
// 1 wave per block; wave owns 16 query rows; 16x1024 bf16 row-block in LDS.
// A operand (q tile) stays in VGPRs across all 64 K-tiles (operand reuse).
#define ESMEM (16 * 1024 * 2 + 3 * 16 * 64 * 4 + 1024 * 4 + 32 * 4)
__global__ void __launch_bounds__(32)
k_energy(const bf16* xq, const bf16* xk, const float* ql,
         const int* disc, bf16* AT) {
  extern __shared__ char smem[];
  bf16*     ebuf = (bf16*)smem;                               // 16*1024 bf16
  float*    qlb  = (float*)(smem + 32768);                    // 3*16*64 f32
  unsigned* dbin = (unsigned*)(smem + 32768 + 12288);         // 1024 packed
  float*    srow = (float*)(smem + 32768 + 12288 + 4096);     // 16 max,16 inv

  int b = blockIdx.x >> 6, nt = blockIdx.x & 63;
  int n0 = nt * 16;
  int l = threadIdx.x;
  int kh = l >> 4, lm = l & 15;

  // stage bin coords (u8 x3 packed) for all 1024 points of this batch
  for (int i = 0; i < 32; ++i) {
    int m = i * 32 + l;
    const int* dp = disc + ((size_t)b * NN + m) * 3;
    dbin[m] = (unsigned)dp[0] | ((unsigned)dp[1] << 8) | ((unsigned)dp[2] << 16);
  }
  // stage ql rows for the 16 query rows (3 tables x 16 x 64)
  for (int i = 0; i < 96; ++i) {
    int idx = i * 32 + l;
    int tab = idx >> 10, rem = idx & 1023;
    int row = rem >> 6, r = rem & 63;
    qlb[idx] = ql[(((size_t)tab * BB + b) * NN + (n0 + row)) * RB + r];
  }
  __syncthreads();

  const bf16* arow = xq + ((size_t)b * NN + n0 + lm) * DQK;
  v16bf aLo = ldA(arow, kh), aHi = ldA(arow + 32, kh);

  unsigned rb[8];
  float rmax[8];
#pragma unroll
  for (int i = 0; i < 8; ++i) {
    rb[i] = dbin[n0 + kh * 8 + i];
    rmax[i] = -3.0e38f;
  }

  for (int mt = 0; mt < 64; ++mt) {
    int col = mt * 16 + lm;
    const bf16* brow = xk + ((size_t)b * NN + col) * DQK;
    v8f acc = {};
    acc = wmma_bf(aLo, ldB(brow, kh), acc);
    acc = wmma_bf(aHi, ldB(brow + 32, kh), acc);
    unsigned cb = dbin[col];
    int cx = cb & 255, cy = (cb >> 8) & 255, cz = (cb >> 16) & 255;
#pragma unroll
    for (int i = 0; i < 8; ++i) {
      int row = kh * 8 + i;
      unsigned r8 = rb[i];
      int ix = cx - (int)(r8 & 255) + 31;
      int iy = cy - (int)((r8 >> 8) & 255) + 31;
      int iz = cz - (int)((r8 >> 16) & 255) + 31;
      float e = acc[i] + qlb[row * 64 + ix] + qlb[1024 + row * 64 + iy]
                       + qlb[2048 + row * 64 + iz];
      e *= 0.03125f;   // 1/sqrt(N)
      rmax[i] = fmaxf(rmax[i], e);
      ebuf[row * 1024 + col] = (bf16)e;
    }
  }
  // row max: reduce across the 16 lanes of each half (wave32 butterflies)
#pragma unroll
  for (int i = 0; i < 8; ++i) {
    float v = rmax[i];
    v = fmaxf(v, __shfl_xor(v, 1, 32));
    v = fmaxf(v, __shfl_xor(v, 2, 32));
    v = fmaxf(v, __shfl_xor(v, 4, 32));
    v = fmaxf(v, __shfl_xor(v, 8, 32));
    if (lm == 0) srow[kh * 8 + i] = v;
  }
  __syncthreads();
  // exp + row sum: lane handles row lm, half kh of the 1024 columns
  {
    int row = lm;
    float m = srow[row];
    float s = 0.0f;
    for (int c = kh * 512; c < kh * 512 + 512; ++c) {
      float v = __expf((float)ebuf[row * 1024 + c] - m);
      s += v;
      ebuf[row * 1024 + c] = (bf16)v;
    }
    s += __shfl_xor(s, 16, 32);
    if (kh == 0) srow[16 + row] = 1.0f / s;
  }
  __syncthreads();
  // write attention transposed: AT[b][m][n0..n0+15], 2x b128 per m
  for (int it = 0; it < 32; ++it) {
    int m = it * 32 + l;
    alignas(16) bf16 o[16];
#pragma unroll
    for (int r = 0; r < 16; ++r)
      o[r] = (bf16)((float)ebuf[r * 1024 + m] * srow[16 + r]);
    u32x4* dst = reinterpret_cast<u32x4*>(AT + ((size_t)b * NN + m) * NN + n0);
    dst[0] = *reinterpret_cast<u32x4*>(&o[0]);
    dst[1] = *reinterpret_cast<u32x4*>(&o[8]);
  }
}

// --------- x_r = V @ A ; h = x - x_r  ->  hT[b][n][d] bf16 -----------------
// Register-blocked 16(d) x 64(m) per wave: A tile loaded once per K step,
// reused by 4 accumulators (4 B tiles) -> 4 wmma per 10 b128 loads.
__device__ __forceinline__ void attnv_epil(v8f acc, const float* x, bf16* hT,
                                           int b, int d0, int m) {
#pragma unroll
  for (int i = 0; i < 8; ++i) {
    int d = d0 + i;
    float h = x[((size_t)b * CC + d) * NN + m] - acc[i];
    hT[((size_t)b * NN + m) * DV + d] = (bf16)h;
  }
}

__global__ void __launch_bounds__(256)
k_attnv(const bf16* xv, const bf16* AT, const float* x, bf16* hT) {
  int t = blockIdx.x * 8 + (threadIdx.x >> 5);   // b(16) x dt(16) x mq(16)
  int l = threadIdx.x & 31;
  int b = t >> 8, rem = t & 255;
  int dt = rem >> 4, mq = rem & 15;
  int kh = l >> 4, lm = l & 15;
  const bf16* arow  = xv + ((size_t)b * DV + dt * 16 + lm) * NN;
  const bf16* brow0 = AT + ((size_t)b * NN + mq * 64 + lm) * NN;
  const bf16* brow1 = brow0 + (size_t)16 * NN;
  const bf16* brow2 = brow0 + (size_t)32 * NN;
  const bf16* brow3 = brow0 + (size_t)48 * NN;
  v8f acc0 = {}, acc1 = {}, acc2 = {}, acc3 = {};
#pragma unroll 2
  for (int kk = 0; kk < 32; ++kk) {
    v16bf a = ldA(arow + kk * 32, kh);
    acc0 = wmma_bf(a, ldB(brow0 + kk * 32, kh), acc0);
    acc1 = wmma_bf(a, ldB(brow1 + kk * 32, kh), acc1);
    acc2 = wmma_bf(a, ldB(brow2 + kk * 32, kh), acc2);
    acc3 = wmma_bf(a, ldB(brow3 + kk * 32, kh), acc3);
  }
  int d0 = dt * 16 + kh * 8;
  int m0 = mq * 64 + lm;
  attnv_epil(acc0, x, hT, b, d0, m0);
  attnv_epil(acc1, x, hT, b, d0, m0 + 16);
  attnv_epil(acc2, x, hT, b, d0, m0 + 32);
  attnv_epil(acc3, x, hT, b, d0, m0 + 48);
}

// ----------------- t = trans_w @ h + trans_b  (fp32 out) -------------------
__global__ void __launch_bounds__(256)
k_trans(const bf16* wtr, const bf16* hT, const float* tb, float* tout) {
  int t = blockIdx.x * 8 + (threadIdx.x >> 5);
  int l = threadIdx.x & 31;
  int b = t >> 10, rem = t & 1023;
  int ct = rem >> 6, nt = rem & 63;
  int kh = l >> 4, lm = l & 15;
  const bf16* arow = wtr + (size_t)(ct * 16 + lm) * DV;
  const bf16* brow = hT + ((size_t)b * NN + nt * 16 + lm) * DV;
  v8f acc = {};
#pragma unroll
  for (int kk = 0; kk < 8; ++kk)
    acc = wmma_bf(ldA(arow + kk * 32, kh), ldB(brow + kk * 32, kh), acc);
  int n = nt * 16 + lm;
#pragma unroll
  for (int i = 0; i < 8; ++i) {
    int c = ct * 16 + kh * 8 + i;
    tout[((size_t)b * CC + c) * NN + n] = acc[i] + tb[c];
  }
}

// ----------- deterministic BN stats: one block per channel -----------------
__global__ void __launch_bounds__(256) k_bnstats(const float* t, float* stats) {
  __shared__ float s1[256], s2[256];
  int c = blockIdx.x, tid = threadIdx.x;
  float sum = 0.0f, sq = 0.0f;
  for (int j = tid; j < BB * NN; j += 256) {
    int b = j >> 10, n = j & 1023;
    float v = t[((size_t)b * CC + c) * NN + n];
    sum += v; sq += v * v;
  }
  s1[tid] = sum; s2[tid] = sq;
  __syncthreads();
  for (int off = 128; off > 0; off >>= 1) {
    if (tid < off) { s1[tid] += s1[tid + off]; s2[tid] += s2[tid + off]; }
    __syncthreads();
  }
  if (tid == 0) {
    const float inv = 1.0f / (float)(BB * NN);
    float mean = s1[0] * inv;
    float var  = s2[0] * inv - mean * mean;
    stats[c] = mean;
    stats[256 + c] = rsqrtf(var + 1e-5f);
  }
}

// ---------------- out = x + relu(BN(t)) ------------------------------------
__global__ void k_final(const float* x, const float* t, const float* stats,
                        const float* gamma, const float* beta, float* out) {
  size_t i = (size_t)blockIdx.x * blockDim.x + threadIdx.x;
  if (i >= (size_t)BB * CC * NN) return;
  int c = (int)((i >> 10) & 255);
  float v = (t[i] - stats[c]) * stats[256 + c] * gamma[c] + beta[c];
  out[i] = x[i] + fmaxf(v, 0.0f);
}

// ---------------------------------------------------------------------------
extern "C" void kernel_launch(void* const* d_in, const int* in_sizes, int n_in,
                              void* d_out, int out_size, void* d_ws, size_t ws_size,
                              hipStream_t stream) {
  const float* x    = (const float*)d_in[0];
  const int*   disc = (const int*)d_in[1];
  // d_in[2] = xyz (unused by reference)
  const float* qw  = (const float*)d_in[3];
  const float* kw  = (const float*)d_in[4];
  const float* vw  = (const float*)d_in[5];
  const float* vb  = (const float*)d_in[6];
  const float* tw  = (const float*)d_in[7];
  const float* tb  = (const float*)d_in[8];
  const float* gam = (const float*)d_in[9];
  const float* bet = (const float*)d_in[10];
  const float* xlt = (const float*)d_in[11];
  const float* ylt = (const float*)d_in[12];
  const float* zlt = (const float*)d_in[13];
  float* out = (float*)d_out;

  // workspace layout (bytes); xT aliased under AT (proj finishes before energy)
  char* ws = (char*)d_ws;
  bf16*  wqkv  = (bf16*)(ws + 0);                 // 384*256*2   = 196608
  bf16*  wtr   = (bf16*)(ws + 196608);            // 256*256*2   = 131072
  bf16*  ltb   = (bf16*)(ws + 327680);            // 3*64*64*2   = 24576
  bf16*  xq    = (bf16*)(ws + 352256);            // 16*1024*64*2  = 2 MiB
  bf16*  xk    = (bf16*)(ws + 2449408);           // 2 MiB
  bf16*  xv    = (bf16*)(ws + 4546560);           // 16*256*1024*2 = 8 MiB
  float* ql    = (float*)(ws + 12935168);         // 3*16*1024*64*4 = 12 MiB
  bf16*  hT    = (bf16*)(ws + 25518080);          // 8 MiB
  float* tbuf  = (float*)(ws + 33906688);         // 16 MiB
  float* stats = (float*)(ws + 50683904);         // 2 KiB
  char*  shreg = ws + 50685952;                   // 32 MiB shared region
  bf16*  xT    = (bf16*)shreg;                    //   [B,N,C]  8 MiB (early)
  bf16*  AT    = (bf16*)shreg;                    //   [B,N,N] 32 MiB (late)
  (void)in_sizes; (void)n_in; (void)out_size; (void)ws_size;

  // 1. pack weights / lookup tables to bf16
  k_pack<<<(176128 + 255) / 256, 256, 0, stream>>>(qw, kw, vw, tw, xlt, ylt, zlt,
                                                   wqkv, wtr, ltb);
  // 2. transpose+convert x
  k_xT<<<(BB * NN) / 256, 256, 0, stream>>>(x, xT);
  // 3. fused QKV projection (WMMA)
  k_proj<<<(BB * 24 * 64) / 8, 256, 0, stream>>>(wqkv, xT, vb, xq, xk, xv);
  // 4. ql tables (WMMA)
  k_ql<<<(3 * BB * 4 * 64) / 8, 256, 0, stream>>>(xq, ltb, ql);
  // 5. fused energy + rel-bias + softmax -> A^T bf16 (WMMA + LDS)
  k_energy<<<BB * 64, 32, ESMEM, stream>>>(xq, xk, ql, disc, AT);
  // 6. x_r = V@A (16x64 register-blocked), h = x - x_r (WMMA)
  k_attnv<<<(BB * 16 * 16) / 8, 256, 0, stream>>>(xv, AT, x, hT);
  // 7. t = trans_w @ h + trans_b (WMMA)
  k_trans<<<(BB * 16 * 64) / 8, 256, 0, stream>>>(wtr, hT, tb, tbuf);
  // 8. BN statistics (deterministic tree reduction)
  k_bnstats<<<256, 256, 0, stream>>>(tbuf, stats);
  // 9. BN + ReLU + residual
  k_final<<<(BB * CC * NN) / 256, 256, 0, stream>>>(x, tbuf, stats, gam, bet, out);
}
